// GraphAttentionLayer_28406913696402
// MI455X (gfx1250) — compile-verified, matrix operand order checked by reading, not run on hardware
//
#include <hip/hip_runtime.h>
#include <hip/hip_bf16.h>
#include <math.h>

#define NN 4096
#define DD 256
#define HH 8
#define HD 32
#define FF 1024   // 4*D

typedef __attribute__((ext_vector_type(16))) _Float16 v16h;
typedef __attribute__((ext_vector_type(8)))  float    v8f;

#define WMMA16(a, b, c) \
  __builtin_amdgcn_wmma_f32_16x16x32_f16(false, (a), false, (b), (short)0, (c), false, false)

// ---------------------------------------------------------------------------
// A/B operand loader for v_wmma_f32_16x16x32_f16.
// A operand, k order chosen here: lane L holds row (L%16), K-half (L/16)*16, 16 halves
// contiguous.  The identical lane pattern loads a B-operand from a matrix
// stored "output-column major over K" (i.e. transposed weights / K-matrix /
// transposed V), where lane L holds output column (L%16).
// ---------------------------------------------------------------------------
static __device__ __forceinline__ v16h ld_tile16x32(const _Float16* __restrict__ base,
                                                    int ld, int row0, int k0) {
  const int lane = threadIdx.x & 31;
  const _Float16* p = base + (size_t)(row0 + (lane & 15)) * ld + k0 + ((lane >> 4) << 4);
  return *(const v16h*)p;
}

// ---------------------------------------------------------------------------
// Mask bitset construction (N*N bits = 2 MB)
// ---------------------------------------------------------------------------
__global__ void k_zero_mask(unsigned* __restrict__ m, int words) {
  int i = blockIdx.x * blockDim.x + threadIdx.x;
  if (i < words) m[i] = 0u;
}

__global__ void k_edge_mask(const int* __restrict__ ei, unsigned* __restrict__ m, int E) {
  int e = blockIdx.x * blockDim.x + threadIdx.x;
  if (e >= E) return;
  unsigned s = (unsigned)ei[e];
  unsigned d = (unsigned)ei[E + e];
  unsigned idx = s * (unsigned)NN + d;
  atomicOr(&m[idx >> 5], 1u << (idx & 31));
}

__global__ void k_diag_mask(unsigned* __restrict__ m) {
  int i = blockIdx.x * blockDim.x + threadIdx.x;
  if (i < NN) {
    unsigned idx = (unsigned)i * (unsigned)NN + (unsigned)i;
    atomicOr(&m[idx >> 5], 1u << (idx & 31));
  }
}

// ---------------------------------------------------------------------------
// Weight convert + transpose: f32 W[K][Nout] -> f16 Wt[Nout][K]
// ---------------------------------------------------------------------------
__global__ void k_wt(const float* __restrict__ W, _Float16* __restrict__ Wt,
                     int K, int Nout) {
  int i = blockIdx.x * blockDim.x + threadIdx.x;
  if (i >= K * Nout) return;
  int k = i / Nout;
  int n = i - k * Nout;
  Wt[(size_t)n * K + k] = (_Float16)W[i];
}

// ---------------------------------------------------------------------------
// LayerNorm over D=256, one 256-thread block per row, f16 output
// ---------------------------------------------------------------------------
__global__ __launch_bounds__(256)
void k_layernorm(const float* __restrict__ x, const float* __restrict__ g,
                 const float* __restrict__ b, _Float16* __restrict__ out) {
  __shared__ float red[256];
  const int row = blockIdx.x, t = threadIdx.x;
  float v = x[(size_t)row * DD + t];
  red[t] = v;
  __syncthreads();
  for (int s = 128; s > 0; s >>= 1) { if (t < s) red[t] += red[t + s]; __syncthreads(); }
  float mu = red[0] * (1.0f / DD);
  __syncthreads();
  float d = v - mu;
  red[t] = d * d;
  __syncthreads();
  for (int s = 128; s > 0; s >>= 1) { if (t < s) red[t] += red[t + s]; __syncthreads(); }
  float var = red[0] * (1.0f / DD);
  float rs = rsqrtf(var + 1e-6f);
  out[(size_t)row * DD + t] = (_Float16)(d * rs * g[t] + b[t]);
}

// ---------------------------------------------------------------------------
// WMMA GEMM: C[M][Nout] = A[M][K](f16) * Wt[Nout][K](f16)^T + bias
// 4 waves/block, each wave one 16x16 tile; optional relu / residual /
// f32 store / f16 store / f16 transposed store.
// ---------------------------------------------------------------------------
template <bool RELU>
__global__ __launch_bounds__(128)
void k_gemm(const _Float16* __restrict__ A, const _Float16* __restrict__ Wt,
            const float* __restrict__ bias, int M, int K, int Nout,
            float* __restrict__ outF32, const float* __restrict__ resid,
            _Float16* __restrict__ outF16, _Float16* __restrict__ outF16T) {
  const int lane = threadIdx.x & 31;
  const int wv   = threadIdx.x >> 5;
  const int row0 = blockIdx.x * 16;
  const int col0 = blockIdx.y * 64 + wv * 16;

  v8f acc = {};
  for (int k0 = 0; k0 < K; k0 += 32) {
    v16h a = ld_tile16x32(A,  K, row0, k0);
    v16h b = ld_tile16x32(Wt, K, col0, k0);
    acc = WMMA16(a, b, acc);
  }

  const int n  = lane & 15;
  const int mb = (lane >> 4) << 3;
  const float bv = bias[col0 + n];
#pragma unroll
  for (int r = 0; r < 8; ++r) {
    const int row = row0 + mb + r;
    float v = acc[r] + bv;
    if (RELU) v = fmaxf(v, 0.0f);
    if (resid)   v += resid[(size_t)row * Nout + col0 + n];
    if (outF32)  outF32[(size_t)row * Nout + col0 + n] = v;
    if (outF16)  outF16[(size_t)row * Nout + col0 + n] = (_Float16)v;
    if (outF16T) outF16T[(size_t)(col0 + n) * M + row] = (_Float16)v;
  }
}

// ---------------------------------------------------------------------------
// Flash attention: one wave per (16-row block, head).
// Q,K row-major [N][256]; Vt transposed [256][N]; ctx f16 [N][256].
// ---------------------------------------------------------------------------
__global__ __launch_bounds__(32)
void k_attn(const _Float16* __restrict__ Q, const _Float16* __restrict__ Km,
            const _Float16* __restrict__ Vt, const unsigned* __restrict__ mbits,
            _Float16* __restrict__ ctx) {
  __shared__ __align__(32) _Float16 Psh[16 * 32];
  const int lane = threadIdx.x & 31;
  const int r0   = blockIdx.x * 16;
  const int h    = blockIdx.y;
  const int n    = lane & 15;
  const int hi   = lane >> 4;
  const int mb   = hi << 3;
  const float scale = 0.17677669529663687f;  // 1/sqrt(HD)

  const v16h qa = ld_tile16x32(Q, DD, r0, h * HD);

  v8f o0 = {}, o1 = {};
  float mrow[8], lrow[8];
#pragma unroll
  for (int r = 0; r < 8; ++r) { mrow[r] = -INFINITY; lrow[r] = 0.0f; }

  for (int c0 = 0; c0 < NN; c0 += 32) {
    v16h kb0 = ld_tile16x32(Km, DD, c0,      h * HD);
    v16h kb1 = ld_tile16x32(Km, DD, c0 + 16, h * HD);
    v8f z = {};
    v8f s0 = WMMA16(qa, kb0, z);
    v8f s1 = WMMA16(qa, kb1, z);

#pragma unroll
    for (int r = 0; r < 8; ++r) {
      const int row = r0 + mb + r;
      const unsigned w = mbits[(((size_t)row * NN) + (size_t)c0) >> 5];
      const bool b0 = (w >> n) & 1u;
      const bool b1 = (w >> (16 + n)) & 1u;
      const float t0 = b0 ? s0[r] * scale : -INFINITY;
      const float t1 = b1 ? s1[r] * scale : -INFINITY;

      float rmax = fmaxf(t0, t1);
#pragma unroll
      for (int off = 8; off > 0; off >>= 1)
        rmax = fmaxf(rmax, __shfl_xor(rmax, off, 16));

      const float mnew  = fmaxf(mrow[r], rmax);
      const float alpha = (mrow[r] == -INFINITY) ? 0.0f : __expf(mrow[r] - mnew);
      const float p0 = b0 ? __expf(t0 - mnew) : 0.0f;
      const float p1 = b1 ? __expf(t1 - mnew) : 0.0f;

      float ps = p0 + p1;
#pragma unroll
      for (int off = 8; off > 0; off >>= 1)
        ps += __shfl_xor(ps, off, 16);

      lrow[r] = lrow[r] * alpha + ps;
      mrow[r] = mnew;
      o0[r] *= alpha;
      o1[r] *= alpha;
      Psh[(mb + r) * 32 + n]      = (_Float16)p0;
      Psh[(mb + r) * 32 + 16 + n] = (_Float16)p1;
    }
    __syncthreads();
    v16h pa  = *(const v16h*)&Psh[(lane & 15) * 32 + (hi << 4)];
    v16h vb0 = ld_tile16x32(Vt, NN, h * HD,      c0);
    v16h vb1 = ld_tile16x32(Vt, NN, h * HD + 16, c0);
    o0 = WMMA16(pa, vb0, o0);
    o1 = WMMA16(pa, vb1, o1);
    __syncthreads();
  }

#pragma unroll
  for (int r = 0; r < 8; ++r) {
    const float inv = 1.0f / lrow[r];
    const int row = r0 + mb + r;
    ctx[(size_t)row * DD + h * HD + n]      = (_Float16)(o0[r] * inv);
    ctx[(size_t)row * DD + h * HD + 16 + n] = (_Float16)(o1[r] * inv);
  }
}

// ---------------------------------------------------------------------------
extern "C" void kernel_launch(void* const* d_in, const int* in_sizes, int n_in,
                              void* d_out, int out_size, void* d_ws, size_t ws_size,
                              hipStream_t stream) {
  (void)n_in; (void)out_size; (void)ws_size;
  const float* x   = (const float*)d_in[0];
  const int*   ei  = (const int*)d_in[1];
  const float* Wq  = (const float*)d_in[2];  const float* bq  = (const float*)d_in[3];
  const float* Wk  = (const float*)d_in[4];  const float* bk  = (const float*)d_in[5];
  const float* Wv  = (const float*)d_in[6];  const float* bv  = (const float*)d_in[7];
  const float* Wo  = (const float*)d_in[8];  const float* bo  = (const float*)d_in[9];
  const float* W1  = (const float*)d_in[10]; const float* bf1 = (const float*)d_in[11];
  const float* W2  = (const float*)d_in[12]; const float* bf2 = (const float*)d_in[13];
  const float* g1  = (const float*)d_in[14]; const float* be1 = (const float*)d_in[15];
  const float* g2  = (const float*)d_in[16]; const float* be2 = (const float*)d_in[17];
  float* out = (float*)d_out;
  const int E = in_sizes[1] / 2;

  // workspace carve-out
  size_t off = 0;
  char* base = (char*)d_ws;
  auto carve = [&](size_t bytes) -> void* {
    void* p = base + off;
    off += (bytes + 255) & ~(size_t)255;
    return p;
  };
  unsigned* mask  = (unsigned*)carve((size_t)NN * NN / 8);       // 2 MB
  _Float16* xln   = (_Float16*)carve((size_t)NN * DD * 2);
  _Float16* qb    = (_Float16*)carve((size_t)NN * DD * 2);
  _Float16* kbuf  = (_Float16*)carve((size_t)NN * DD * 2);
  _Float16* vt    = (_Float16*)carve((size_t)NN * DD * 2);       // transposed V [DD][NN]
  _Float16* ctx   = (_Float16*)carve((size_t)NN * DD * 2);
  float*    y     = (float*)   carve((size_t)NN * DD * 4);       // attn out + residual
  _Float16* yln   = (_Float16*)carve((size_t)NN * DD * 2);
  _Float16* hbuf  = (_Float16*)carve((size_t)NN * FF * 2);       // 8 MB
  _Float16* wqt   = (_Float16*)carve((size_t)DD * DD * 2);
  _Float16* wkt   = (_Float16*)carve((size_t)DD * DD * 2);
  _Float16* wvt   = (_Float16*)carve((size_t)DD * DD * 2);
  _Float16* wot   = (_Float16*)carve((size_t)DD * DD * 2);
  _Float16* w1t   = (_Float16*)carve((size_t)FF * DD * 2);
  _Float16* w2t   = (_Float16*)carve((size_t)DD * FF * 2);

  // 1. mask bitset
  const int words = NN * NN / 32;
  k_zero_mask<<<(words + 255) / 256, 256, 0, stream>>>(mask, words);
  k_edge_mask<<<(E + 255) / 256, 256, 0, stream>>>(ei, mask, E);
  k_diag_mask<<<(NN + 255) / 256, 256, 0, stream>>>(mask);

  // 2. weight convert + transpose to f16
  k_wt<<<(DD * DD + 255) / 256, 256, 0, stream>>>(Wq, wqt, DD, DD);
  k_wt<<<(DD * DD + 255) / 256, 256, 0, stream>>>(Wk, wkt, DD, DD);
  k_wt<<<(DD * DD + 255) / 256, 256, 0, stream>>>(Wv, wvt, DD, DD);
  k_wt<<<(DD * DD + 255) / 256, 256, 0, stream>>>(Wo, wot, DD, DD);
  k_wt<<<(DD * FF + 255) / 256, 256, 0, stream>>>(W1, w1t, DD, FF);
  k_wt<<<(FF * DD + 255) / 256, 256, 0, stream>>>(W2, w2t, FF, DD);

  // 3. LN1
  k_layernorm<<<NN, 256, 0, stream>>>(x, g1, be1, xln);

  // 4. QKV projections (WMMA): Q,K as f16 row-major; V stored transposed
  dim3 gdim(NN / 16, DD / 64);
  k_gemm<false><<<gdim, 128, 0, stream>>>(xln, wqt, bq, NN, DD, DD,
                                          nullptr, nullptr, qb, nullptr);
  k_gemm<false><<<gdim, 128, 0, stream>>>(xln, wkt, bk, NN, DD, DD,
                                          nullptr, nullptr, kbuf, nullptr);
  k_gemm<false><<<gdim, 128, 0, stream>>>(xln, wvt, bv, NN, DD, DD,
                                          nullptr, nullptr, nullptr, vt);

  // 5. masked flash attention (WMMA QK^T and PV)
  k_attn<<<dim3(NN / 16, HH), 32, 0, stream>>>(qb, kbuf, vt, mask, ctx);

  // 6. output projection + residual -> y (f32)
  k_gemm<false><<<gdim, 128, 0, stream>>>(ctx, wot, bo, NN, DD, DD,
                                          y, x, nullptr, nullptr);

  // 7. LN2
  k_layernorm<<<NN, 256, 0, stream>>>(y, g2, be2, yln);

  // 8. FFN: relu(yln @ W1 + bf1) -> hbuf (f16); hbuf @ W2 + bf2 + y -> out
  k_gemm<true><<<dim3(NN / 16, FF / 64), 128, 0, stream>>>(yln, w1t, bf1, NN, DD, FF,
                                                           nullptr, nullptr, hbuf, nullptr);
  k_gemm<false><<<dim3(NN / 16, DD / 64), 128, 0, stream>>>(hbuf, w2t, bf2, NN, FF, DD,
                                                            out, y, nullptr, nullptr);
}
